// PSMNet_15255723835827
// MI455X (gfx1250) — matrix operand order repaired and checked by reading, not compiled, not run on hardware
//
#include <hip/hip_runtime.h>
#include <hip/hip_bf16.h>

// ---- problem constants (from reference) ----
#define TT   4
#define CC   64
#define HHI  480
#define WWI  640
#define PP   (HHI * WWI)      // 307200 pixels / frame
#define MW   256
#define HWC  (MW * MW)        // 65536 map cells
#define MEMN 256              // GRU hidden
#define KC   320              // concat K: 64 (x) + 256 (h)
#define NC   1024             // gate cols: r(256) z(256) i_n(256) h_n(256)

typedef __attribute__((ext_vector_type(8)))  _Float16 v8h;
typedef __attribute__((ext_vector_type(16))) _Float16 v16h;
typedef __attribute__((ext_vector_type(8)))  float    v8f;

union V16 { v16h v; v8h h[2]; };

// orderable encoding of float for u64 atomicMax key
__device__ __forceinline__ unsigned int fenc(float f) {
    unsigned int u = __float_as_uint(f);
    return (u & 0x80000000u) ? ~u : (u | 0x80000000u);
}
__device__ __forceinline__ float fdec(unsigned int k) {
    unsigned int u = (k & 0x80000000u) ? (k ^ 0x80000000u) : ~k;
    return __uint_as_float(u);
}

// ---------- one-time weight prep: Wcat f16 [NC][KC], biascat f32 [NC] ----------
// rows 0..511   : [W_ih(row j) | W_hh(row j)]        (combined r,z pre-activation)
// rows 512..767 : [W_ih(row j) | 0]                  (i_n)
// rows 768..1023: [0           | W_hh(row j-256)]    (h_n)
__global__ void k_prep(const float* __restrict__ W_ih, const float* __restrict__ W_hh,
                       const float* __restrict__ b_ih, const float* __restrict__ b_hh,
                       _Float16* __restrict__ Wcat, float* __restrict__ biascat)
{
    int i = blockIdx.x * 256 + threadIdx.x;
    if (i < NC * KC) {
        int j = i / KC, k = i % KC;
        float v;
        if (j < 512)       v = (k < 64) ? W_ih[j * 64 + k] : W_hh[j * 256 + (k - 64)];
        else if (j < 768)  v = (k < 64) ? W_ih[j * 64 + k] : 0.0f;
        else               v = (k < 64) ? 0.0f : W_hh[(j - 256) * 256 + (k - 64)];
        Wcat[i] = (_Float16)v;
    }
    if (i < NC) {
        float b;
        if (i < 512)      b = b_ih[i] + b_hh[i];
        else if (i < 768) b = b_ih[i];
        else              b = b_hh[i - 256];
        biascat[i] = b;
    }
}

// ---------- zero state / Xcat / observed / height_map ----------
__global__ void k_init(float* __restrict__ state, unsigned int* __restrict__ xcat32,
                       float* __restrict__ obs, float* __restrict__ hmap)
{
    int i = blockIdx.x * 256 + threadIdx.x;            // 16777216 threads
    state[i] = 0.0f;
    if (i < (HWC * KC) / 2) xcat32[i] = 0u;
    if (i < HWC) { obs[i] = 0.0f; hmap[i] = 0.0f; }
}

__global__ void k_clearkeys(unsigned long long* __restrict__ cellkey)
{
    int i = blockIdx.x * 256 + threadIdx.x;
    if (i < HWC) cellkey[i] = 0ull;
}

// ---------- scatter: per-cell (max height, min pixel) via packed u64 atomicMax ----------
__global__ void k_scatter(const int* __restrict__ proj, const unsigned char* __restrict__ mask,
                          const float* __restrict__ hgt, unsigned long long* __restrict__ cellkey)
{
    int p = blockIdx.x * 256 + threadIdx.x;
    if (p >= PP) return;
    if (mask[p]) return;                               // outlier -> skip
    int x = proj[2 * p + 0];
    int y = proj[2 * p + 1];
    int idx = y * MW + x;
    float hv = hgt[p] + 1000.0f;
    unsigned long long key =
        ((unsigned long long)fenc(hv) << 32) | (unsigned int)(~(unsigned int)p);
    atomicMax(&cellkey[idx], key);
}

// ---------- decode winners, update height map / observed ----------
__global__ void k_decode(const unsigned long long* __restrict__ cellkey,
                         float* __restrict__ hmap, float* __restrict__ obs,
                         unsigned int* __restrict__ updated, int* __restrict__ winpix)
{
    int c = blockIdx.x * 256 + threadIdx.x;
    if (c >= HWC) return;
    unsigned long long key = cellkey[c];
    unsigned int pix = ~(unsigned int)(key & 0xFFFFFFFFull);
    if (pix > (unsigned int)(PP - 1)) pix = PP - 1;    // clamp like reference
    bool has = (key != 0ull);
    float hv = fdec((unsigned int)(key >> 32));
    bool upd = has && (hv > hmap[c]);
    if (upd) { hmap[c] = hv; obs[c] = 1.0f; }
    updated[c] = upd ? 1u : 0u;
    winpix[c] = (int)pix;
}

// ---------- gather winner features into Xcat[:, 0:64] as f16 ----------
__global__ void k_gather(const float* __restrict__ feat, const int* __restrict__ winpix,
                         _Float16* __restrict__ Xcat)
{
    int i = blockIdx.x * 256 + threadIdx.x;            // HWC*64 threads
    int cell = i >> 6;
    int c = i & 63;
    Xcat[cell * KC + c] = (_Float16)feat[(size_t)c * PP + winpix[cell]];
}

// ---------- fused WMMA GEMM + GRU gates ----------
// block = 512 threads = 16 waves; each block owns one M-tile (16 cells).
// wave w computes N-tiles [4w, 4w+4); K=320 in 10 chunks of 32 via v_wmma_f32_16x16x32_f16.
__global__ __launch_bounds__(512)
void k_gru(_Float16* __restrict__ Xcat, const _Float16* __restrict__ Wcat,
           const float* __restrict__ biascat, const unsigned int* __restrict__ updated,
           float* __restrict__ state, float* __restrict__ out_mem, int write_out)
{
    __shared__ float g[16][NC];                        // 64 KB gate pre-activations
    const int tid  = threadIdx.x;
    const int wid  = tid >> 5;
    const int lane = tid & 31;
    const int r    = lane & 15;
    const int kh   = lane >> 4;                        // lane half (wave32)
    const int tile = blockIdx.x;                       // cells tile*16 .. +15

    const _Float16* pA = Xcat + (size_t)(tile * 16 + r) * KC;
    const _Float16* pB0 = Wcat + (size_t)((wid * 4 + 0) * 16 + r) * KC;
    const _Float16* pB1 = Wcat + (size_t)((wid * 4 + 1) * 16 + r) * KC;
    const _Float16* pB2 = Wcat + (size_t)((wid * 4 + 2) * 16 + r) * KC;
    const _Float16* pB3 = Wcat + (size_t)((wid * 4 + 3) * 16 + r) * KC;
    const _Float16* pB[4] = { pB0, pB1, pB2, pB3 };

    v8f acc[4] = {};                                   // zero accumulators

    #pragma unroll
    for (int kb = 0; kb < 10; kb++) {
        // A 16x32 f16 frag: halves [0..7] at K = kb*32 + kh*8, [8..15] at +16
        V16 a;
        a.h[0] = *(const v8h*)(pA + kb * 32 + kh * 8);
        a.h[1] = *(const v8h*)(pA + kb * 32 + 16 + kh * 8);
        #pragma unroll
        for (int i = 0; i < 4; i++) {
            // B 32x16 f16 frag: halves [0..7] at K = kb*32 + kh*8, [8..15] at +16 (row n of W)
            V16 b;
            b.h[0] = *(const v8h*)(pB[i] + kb * 32 + kh * 8);
            b.h[1] = *(const v8h*)(pB[i] + kb * 32 + 16 + kh * 8);
            acc[i] = __builtin_amdgcn_wmma_f32_16x16x32_f16(
                false, a.v, false, b.v, (short)0, acc[i], false, false);
        }
    }

    // spill D tiles to LDS: VGPR gidx -> M = kh*8 + gidx, N = n0 + r
    #pragma unroll
    for (int i = 0; i < 4; i++) {
        int n0 = (wid * 4 + i) * 16;
        #pragma unroll
        for (int gi = 0; gi < 8; gi++)
            g[kh * 8 + gi][n0 + r] = acc[i][gi];
    }
    __syncthreads();

    // gate math: 16 cells x 256 hidden = 4096 items / 512 threads
    #pragma unroll
    for (int it = 0; it < 8; it++) {
        int lin = it * 512 + tid;
        int lr = lin >> 8;
        int j  = lin & 255;
        int cell = tile * 16 + lr;
        float rpre = g[lr][j]       + biascat[j];
        float zpre = g[lr][256 + j] + biascat[256 + j];
        float inn  = g[lr][512 + j] + biascat[512 + j];
        float hnn  = g[lr][768 + j] + biascat[768 + j];
        float rg = 1.0f / (1.0f + __expf(-rpre));
        float zg = 1.0f / (1.0f + __expf(-zpre));
        float ng = tanhf(inn + rg * hnn);
        float h  = state[(size_t)cell * MEMN + j];
        float hnew = (1.0f - zg) * ng + zg * h;
        float outv = updated[cell] ? hnew : h;
        state[(size_t)cell * MEMN + j] = outv;
        Xcat[(size_t)cell * KC + 64 + j] = (_Float16)outv;   // h in f16 for next frame
        if (write_out) out_mem[(size_t)j * HWC + cell] = outv; // [1,mem,H,W] transpose
    }
}

extern "C" void kernel_launch(void* const* d_in, const int* in_sizes, int n_in,
                              void* d_out, int out_size, void* d_ws, size_t ws_size,
                              hipStream_t stream) {
    const float*         features = (const float*)d_in[0];
    const int*           proj     = (const int*)d_in[1];
    const unsigned char* mask     = (const unsigned char*)d_in[2];   // jax bool_
    const float*         heights  = (const float*)d_in[3];
    const float*         W_ih     = (const float*)d_in[6];
    const float*         W_hh     = (const float*)d_in[7];
    const float*         b_ih     = (const float*)d_in[8];
    const float*         b_hh     = (const float*)d_in[9];

    float* out      = (float*)d_out;
    float* out_mem  = out;                         // 256*65536
    float* out_obs  = out + (size_t)MEMN * HWC;    // 65536
    float* out_hmap = out_obs + HWC;               // 65536

    char* ws = (char*)d_ws;
    unsigned long long* cellkey = (unsigned long long*)(ws);                 // 512 KB
    unsigned int*       updated = (unsigned int*)(ws + 524288);              // 256 KB
    int*                winpix  = (int*)(ws + 786432);                       // 256 KB
    float*              state   = (float*)(ws + 1048576);                    // 64 MB
    _Float16*           Xcat    = (_Float16*)(ws + 68157440);                // 40 MB
    _Float16*           Wcat    = (_Float16*)(ws + 110100480);               // 640 KB
    float*              biascat = (float*)(ws + 110755840);                  // 4 KB

    k_prep<<<(NC * KC + 255) / 256, 256, 0, stream>>>(W_ih, W_hh, b_ih, b_hh, Wcat, biascat);
    k_init<<<(HWC * MEMN) / 256, 256, 0, stream>>>(state, (unsigned int*)Xcat, out_obs, out_hmap);

    for (int t = 0; t < TT; t++) {
        k_clearkeys<<<HWC / 256, 256, 0, stream>>>(cellkey);
        k_scatter<<<(PP + 255) / 256, 256, 0, stream>>>(
            proj + (size_t)t * PP * 2, mask + (size_t)t * PP, heights + (size_t)t * PP, cellkey);
        k_decode<<<HWC / 256, 256, 0, stream>>>(cellkey, out_hmap, out_obs, updated, winpix);
        k_gather<<<(HWC * 64) / 256, 256, 0, stream>>>(
            features + (size_t)t * CC * PP, winpix, Xcat);
        k_gru<<<HWC / 16, 512, 0, stream>>>(
            Xcat, Wcat, biascat, updated, state, out_mem, (t == TT - 1) ? 1 : 0);
    }
}
